// DGCN_22127671509732
// MI455X (gfx1250) — matrix operand order repaired and checked by reading, not compiled, not run on hardware
//
#include <hip/hip_runtime.h>
#include <hip/hip_bf16.h>

// ---------------- problem constants ----------------
constexpr int BB = 16;          // batch
constexpr int C  = 64;          // channels
constexpr int NN = 1024;        // nodes
constexpr int T  = 12;          // time
constexpr int NT = NN * T;      // 12288
constexpr int CT = C * T;       // 768
constexpr int TOPK = (NN * 8) / 10;  // 819

typedef __attribute__((ext_vector_type(16))) _Float16 v16h;
typedef __attribute__((ext_vector_type(8)))  _Float16 v8h;
typedef __attribute__((ext_vector_type(8)))  float    v8f;

// ---------------- WMMA helpers (wave32, 16x16x32 f16 -> f32) ----------------
__device__ __forceinline__ v8f wmma16(v16h a, v16h b, v8f c) {
    return __builtin_amdgcn_wmma_f32_16x16x32_f16(
        /*neg_a=*/false, a, /*neg_b=*/false, b,
        /*c_mod=*/(short)0, c, /*reuse_a=*/false, /*reuse_b=*/false);
}

// A fragment: LDS matrix As[m][k] row-major, ld halves per row.
// Lane L holds row m = m0 + (L&15); k base = (L>>4)*8; halves h=0..7 -> k..k+7,
// h=8..15 -> k+16..k+23   (matches 16-bit A 16x32 VGPR layout)
__device__ __forceinline__ v16h load_frag_a(const _Float16* As, int ld,
                                            int m0, int k0, int lane) {
    int m  = m0 + (lane & 15);
    int kb = k0 + ((lane >> 4) << 3);
    const v8h* p0 = (const v8h*)(As + m * ld + kb);
    const v8h* p1 = (const v8h*)(As + m * ld + kb + 16);
    v8h a = *p0, b = *p1;
    return __builtin_shufflevector(a, b, 0,1,2,3,4,5,6,7,8,9,10,11,12,13,14,15);
}

// B fragment: LDS matrix Bs[col][k] (transposed), ld halves per col-row.
// Lane L holds col n = n0 + (L&15); k base = k0 + (L>>4)*8; h=0..7 -> k..k+7, h=8..15 -> k+16..k+23.
__device__ __forceinline__ v16h load_frag_b(const _Float16* Bs, int ld,
                                            int n0, int k0, int lane) {
    int col = n0 + (lane & 15);
    int kb  = k0 + ((lane >> 4) << 3);
    const v8h* p0 = (const v8h*)(Bs + col * ld + kb);
    v8h a = *p0, b = *(const v8h*)(Bs + col * ld + kb + 16);
    return __builtin_shufflevector(a, b, 0,1,2,3,4,5,6,7,8,9,10,11,12,13,14,15);
}

// ---------------- K1: query/value conv1x1 + ReLU ----------------
// query f32 [B][C][NT] ; value -> VR^T f16 [B][CT][NN]
__global__ __launch_bounds__(256) void k1_qv(
    const float* __restrict__ x,
    const float* __restrict__ qw, const float* __restrict__ qb,
    const float* __restrict__ vw, const float* __restrict__ vb,
    float* __restrict__ query, _Float16* __restrict__ vrT)
{
    __shared__ alignas(64) _Float16 Aq[64 * 64];
    __shared__ alignas(64) _Float16 Av[64 * 64];
    __shared__ alignas(64) _Float16 Xs[64 * 64];  // [col][c]
    int blk  = blockIdx.x;
    int b    = blk / (NT / 64);
    int col0 = (blk % (NT / 64)) * 64;
    int tid  = threadIdx.x;
    {
        int r = tid >> 2, cs = (tid & 3) * 16;
        const float* qr = qw + r * 64 + cs;
        const float* vr = vw + r * 64 + cs;
        for (int i = 0; i < 16; ++i) {
            Aq[r * 64 + cs + i] = (_Float16)qr[i];
            Av[r * 64 + cs + i] = (_Float16)vr[i];
        }
        // X transposed stage: row c = r, 16 contiguous cols
        const float* xr = x + ((size_t)b * C + r) * NT + col0 + cs;
        for (int i = 0; i < 16; ++i) Xs[(cs + i) * 64 + r] = (_Float16)xr[i];
    }
    __syncthreads();
    int lane = tid & 31, w = tid >> 5;
    for (int tt = 0; tt < 4; ++tt) {
        int id  = w * 4 + tt;                 // 0..31: 16 Q tiles + 16 V tiles
        const _Float16* A = (id < 16) ? Aq : Av;
        int idx = id & 15;
        int m0 = (idx >> 2) * 16, n0 = (idx & 3) * 16;
        v8f acc = {};
        for (int ks = 0; ks < 2; ++ks) {
            v16h af = load_frag_a(A,  64, m0, ks * 32, lane);
            v16h bf = load_frag_b(Xs, 64, n0, ks * 32, lane);
            acc = wmma16(af, bf, acc);
        }
        int ncol = lane & 15, hi = lane >> 4;
        int colg = col0 + n0 + ncol;
        if (id < 16) {
            for (int r8 = 0; r8 < 8; ++r8) {
                int o = m0 + r8 + hi * 8;
                float vv = acc[r8] + qb[o];
                vv = vv > 0.f ? vv : 0.f;
                query[((size_t)b * C + o) * NT + colg] = vv;
            }
        } else {
            int n = colg / T, t = colg % T;
            for (int r8 = 0; r8 < 8; ++r8) {
                int o = m0 + r8 + hi * 8;
                float vv = acc[r8] + vb[o];
                vv = vv > 0.f ? vv : 0.f;
                vrT[((size_t)b * CT + o * T + t) * NN + n] = (_Float16)vv;
            }
        }
    }
}

// ---------------- K2: qs^T = sum_t query, f16 [B][NN][C] ----------------
__global__ __launch_bounds__(256) void k2_qs(const float* __restrict__ query,
                                             _Float16* __restrict__ qsT)
{
    int blk = blockIdx.x;                 // B * NN/4 blocks
    int b   = blk >> 8;
    int n   = (blk & 255) * 4 + (threadIdx.x >> 6);
    int c   = threadIdx.x & 63;
    const float* p = query + ((size_t)b * C + c) * NT + n * T;
    float s = 0.f;
    for (int t = 0; t < T; ++t) s += p[t];
    qsT[((size_t)b * NN + n) * 64 + c] = (_Float16)s;
}

// ---------------- K3: A1 = relu(qs^T.mem/8), A2 = relu(qs^T.qs/8) ----------------
__global__ __launch_bounds__(256) void k3_adj(
    const _Float16* __restrict__ qsT, const float* __restrict__ memf,
    float* __restrict__ a1, float* __restrict__ a2)
{
    __shared__ alignas(64) _Float16 As[64 * 64];
    __shared__ alignas(64) _Float16 Bs[64 * 64];  // [col][k]
    int blk = blockIdx.x;
    int mx  = blk & 15;
    int my  = (blk >> 4) & 15;
    int mat = (blk >> 8) & 1;
    int b   = blk >> 9;
    int m0g = my * 64, n0g = mx * 64;
    int tid = threadIdx.x;
    {
        int r = tid >> 2, cs = (tid & 3) * 16;
        const _Float16* src = qsT + ((size_t)b * NN + m0g + r) * 64;
        for (int i = 0; i < 16; ++i) As[r * 64 + cs + i] = src[cs + i];
        if (mat == 0) {
            for (int i = 0; i < 16; ++i)
                Bs[r * 64 + cs + i] = (_Float16)memf[(size_t)(cs + i) * NN + n0g + r];
        } else {
            const _Float16* s2 = qsT + ((size_t)b * NN + n0g + r) * 64;
            for (int i = 0; i < 16; ++i) Bs[r * 64 + cs + i] = s2[cs + i];
        }
    }
    __syncthreads();
    int lane = tid & 31, w = tid >> 5;
    float* out = mat ? a2 : a1;
    for (int tt = 0; tt < 2; ++tt) {
        int idx = w * 2 + tt;
        int m0 = (idx >> 2) * 16, n0 = (idx & 3) * 16;
        v8f acc = {};
        for (int ks = 0; ks < 2; ++ks) {
            v16h af = load_frag_a(As, 64, m0, ks * 32, lane);
            v16h bf = load_frag_b(Bs, 64, n0, ks * 32, lane);
            acc = wmma16(af, bf, acc);
        }
        int ncol = lane & 15, hi = lane >> 4;
        int colg = n0g + n0 + ncol;
        for (int r8 = 0; r8 < 8; ++r8) {
            int row = m0g + m0 + r8 + hi * 8;
            float vv = acc[r8] * 0.125f;
            vv = vv > 0.f ? vv : 0.f;
            out[((size_t)b * NN + row) * NN + colg] = vv;
        }
    }
}

// ---------------- K4: softmax+MLP+softmax+topk-mask, adj f16 ----------------
__device__ __forceinline__ float block_max(float v, float* red) {
    int tid = threadIdx.x;
    red[tid] = v; __syncthreads();
    for (int s = 128; s > 0; s >>= 1) {
        if (tid < s) red[tid] = fmaxf(red[tid], red[tid + s]);
        __syncthreads();
    }
    float r = red[0]; __syncthreads();
    return r;
}
__device__ __forceinline__ float block_sum(float v, float* red) {
    int tid = threadIdx.x;
    red[tid] = v; __syncthreads();
    for (int s = 128; s > 0; s >>= 1) {
        if (tid < s) red[tid] += red[tid + s];
        __syncthreads();
    }
    float r = red[0]; __syncthreads();
    return r;
}

__global__ __launch_bounds__(256) void k4_fuse(
    const float* __restrict__ a1, const float* __restrict__ a2,
    const float* __restrict__ fc1w, const float* __restrict__ fc1b,
    const float* __restrict__ fc2w, const float* __restrict__ fc2b,
    _Float16* __restrict__ adjh)
{
    __shared__ float red[256];
    __shared__ unsigned hist[256];
    __shared__ unsigned sPrefix;
    __shared__ int sRemain;
    size_t row = blockIdx.x;              // b*NN + n
    int tid = threadIdx.x;
    const float* r1 = a1 + row * NN;
    const float* r2 = a2 + row * NN;
    float v1[4], v2[4];
    for (int i = 0; i < 4; ++i) { v1[i] = r1[tid * 4 + i]; v2[i] = r2[tid * 4 + i]; }

    // softmax over a1 row
    float m = fmaxf(fmaxf(v1[0], v1[1]), fmaxf(v1[2], v1[3]));
    m = block_max(m, red);
    float sl = 0.f;
    for (int i = 0; i < 4; ++i) { v1[i] = __expf(v1[i] - m); sl += v1[i]; }
    float s = block_sum(sl, red);
    float inv = 1.f / s;
    for (int i = 0; i < 4; ++i) v1[i] *= inv;

    // softmax over a2 row
    m = fmaxf(fmaxf(v2[0], v2[1]), fmaxf(v2[2], v2[3]));
    m = block_max(m, red);
    sl = 0.f;
    for (int i = 0; i < 4; ++i) { v2[i] = __expf(v2[i] - m); sl += v2[i]; }
    s = block_sum(sl, red);
    inv = 1.f / s;
    for (int i = 0; i < 4; ++i) v2[i] *= inv;

    // MLP 2 -> 8 -> 1
    float lv[4];
    float b2 = fc2b[0];
    for (int i = 0; i < 4; ++i) {
        float acc = b2;
        for (int j = 0; j < 8; ++j) {
            float h = fc1w[j * 2 + 0] * v1[i] + fc1w[j * 2 + 1] * v2[i] + fc1b[j];
            h = h > 0.f ? h : 0.f;
            acc += fc2w[j] * h;
        }
        lv[i] = acc;
    }

    // softmax over MLP scores
    m = fmaxf(fmaxf(lv[0], lv[1]), fmaxf(lv[2], lv[3]));
    m = block_max(m, red);
    sl = 0.f;
    for (int i = 0; i < 4; ++i) { lv[i] = __expf(lv[i] - m); sl += lv[i]; }
    s = block_sum(sl, red);
    inv = 1.f / s;
    for (int i = 0; i < 4; ++i) lv[i] *= inv;

    // radix-select the TOPK-th largest (all values > 0 => float bits are ordered)
    if (tid == 0) { sPrefix = 0u; sRemain = TOPK; }
    __syncthreads();
    for (int pass = 3; pass >= 0; --pass) {
        hist[tid] = 0u;
        __syncthreads();
        unsigned pm   = (pass == 3) ? 0u : (0xFFFFFFFFu << ((pass + 1) * 8));
        unsigned pref = sPrefix;
        for (int i = 0; i < 4; ++i) {
            unsigned bits = __float_as_uint(lv[i]);
            if ((bits & pm) == (pref & pm))
                atomicAdd(&hist[(bits >> (pass * 8)) & 255u], 1u);
        }
        __syncthreads();
        if (tid == 0) {
            int rem = sRemain;
            unsigned bsel = 0;
            for (int q = 255; q >= 0; --q) {
                int hc = (int)hist[q];
                if (rem <= hc) { bsel = (unsigned)q; break; }
                rem -= hc;
                if (q == 0) bsel = 0;
            }
            sPrefix |= (bsel << (pass * 8));
            sRemain = rem;
        }
        __syncthreads();
    }
    float thresh = __uint_as_float(sPrefix);

    _Float16* orow = adjh + row * NN;
    for (int i = 0; i < 4; ++i) {
        float v = lv[i];
        orow[tid * 4 + i] = (_Float16)((v >= thresh) ? v : 0.f);
    }
}

// ---------------- K6: AGG = ADJ x VR  (dominant GEMM) ----------------
// Double-buffered CDNA5 async Global->LDS DMA: issue next tile into the
// alternate LDS buffer, wait asynccnt<=2 (previous pair retired, in-order),
// barrier, compute current tile while the DMA fills the other buffer.
__global__ __launch_bounds__(256) void k6_agg(
    const _Float16* __restrict__ adjh, const _Float16* __restrict__ vrT,
    _Float16* __restrict__ aggh)
{
    __shared__ alignas(64) _Float16 As[2][64 * 32];   // [m][k]
    __shared__ alignas(64) _Float16 Bs[2][64 * 32];   // [col][k]
    int blk = blockIdx.x;
    int jx  = blk % 12; blk /= 12;
    int my  = blk % 16;
    int b   = blk / 16;
    int m0g = my * 64, j0 = jx * 64;
    int tid = threadIdx.x, lane = tid & 31, w = tid >> 5;
    int r = tid >> 2, cs = (tid & 3) * 8;

    // uniform per-batch SGPR bases + per-lane 32-bit byte offsets (GVS form)
    const _Float16* baseA = adjh + (size_t)b * NN * NN;
    const _Float16* baseB = vrT  + (size_t)b * CT * NN;
    unsigned offA = (unsigned)(((unsigned)(m0g + r) * NN + cs) * 2u);
    unsigned offB = (unsigned)(((unsigned)(j0  + r) * NN + cs) * 2u);
    // LDS aperture maps flat addr[31:0] directly to the LDS byte offset
    unsigned ldsA0 = (unsigned)(size_t)(&As[0][r * 32 + cs]);
    unsigned ldsA1 = (unsigned)(size_t)(&As[1][r * 32 + cs]);
    unsigned ldsB0 = (unsigned)(size_t)(&Bs[0][r * 32 + cs]);
    unsigned ldsB1 = (unsigned)(size_t)(&Bs[1][r * 32 + cs]);

    // prologue: stage k0 = 0 into buffer 0
    asm volatile("global_load_async_to_lds_b128 %0, %1, %2"
                 :: "v"(ldsA0), "v"(offA), "s"(baseA) : "memory");
    asm volatile("global_load_async_to_lds_b128 %0, %1, %2"
                 :: "v"(ldsB0), "v"(offB), "s"(baseB) : "memory");

    v8f acc0 = {}, acc1 = {};
    int t0 = w * 2;
    int mA0 = ((t0) >> 2) * 16,     nA0 = ((t0) & 3) * 16;
    int mA1 = ((t0 + 1) >> 2) * 16, nA1 = ((t0 + 1) & 3) * 16;
    for (int k0 = 0; k0 < NN; k0 += 32) {
        int cur = (k0 >> 5) & 1;
        if (k0 + 32 < NN) {
            // issue next tile into the other buffer, then drain only the
            // previous pair (ASYNC completions retire in order per wave)
            unsigned kb = (unsigned)((k0 + 32) * 2);
            unsigned la = cur ? ldsA0 : ldsA1;
            unsigned lb = cur ? ldsB0 : ldsB1;
            asm volatile("global_load_async_to_lds_b128 %0, %1, %2"
                         :: "v"(la), "v"(offA + kb), "s"(baseA) : "memory");
            asm volatile("global_load_async_to_lds_b128 %0, %1, %2"
                         :: "v"(lb), "v"(offB + kb), "s"(baseB) : "memory");
            asm volatile("s_wait_asynccnt 0x2" ::: "memory");
        } else {
            asm volatile("s_wait_asynccnt 0x0" ::: "memory");
        }
        __syncthreads();
        {
            const _Float16* Ac = As[cur];
            const _Float16* Bc = Bs[cur];
            v16h af = load_frag_a(Ac, 32, mA0, 0, lane);
            v16h bf = load_frag_b(Bc, 32, nA0, 0, lane);
            acc0 = wmma16(af, bf, acc0);
            af = load_frag_a(Ac, 32, mA1, 0, lane);
            bf = load_frag_b(Bc, 32, nA1, 0, lane);
            acc1 = wmma16(af, bf, acc1);
        }
        __syncthreads();
    }
    int ncol = lane & 15, hi = lane >> 4;
    for (int r8 = 0; r8 < 8; ++r8) {
        int rowA = m0g + mA0 + r8 + hi * 8;
        aggh[((size_t)b * NN + rowA) * CT + j0 + nA0 + ncol] = (_Float16)acc0[r8];
        int rowB = m0g + mA1 + r8 + hi * 8;
        aggh[((size_t)b * NN + rowB) * CT + j0 + nA1 + ncol] = (_Float16)acc1[r8];
    }
}

// ---------------- K7: out = relu(cw x AGG_r + cb), f32 [B][C][NN][T] ----------------
__global__ __launch_bounds__(256) void k7_out(
    const _Float16* __restrict__ aggh,
    const float* __restrict__ cw, const float* __restrict__ cb,
    float* __restrict__ out)
{
    __shared__ alignas(64) _Float16 Aw[64 * 64];
    __shared__ alignas(64) _Float16 Bs[64 * 64];  // [col][c]
    int blk  = blockIdx.x;
    int b    = blk / (NT / 64);
    int col0 = (blk % (NT / 64)) * 64;
    int tid  = threadIdx.x;
    {
        int r = tid >> 2, cs = (tid & 3) * 16;
        const float* wsrc = cw + r * 64 + cs;
        for (int i = 0; i < 16; ++i) Aw[r * 64 + cs + i] = (_Float16)wsrc[i];
        int colg = col0 + r;
        int n = colg / T, t = colg % T;
        const _Float16* asrc = aggh + ((size_t)b * NN + n) * CT + t;
        for (int i = 0; i < 16; ++i) Bs[r * 64 + cs + i] = asrc[(size_t)(cs + i) * T];
    }
    __syncthreads();
    int lane = tid & 31, w = tid >> 5;
    for (int tt = 0; tt < 2; ++tt) {
        int idx = w * 2 + tt;
        int m0 = (idx >> 2) * 16, n0 = (idx & 3) * 16;
        v8f acc = {};
        for (int ks = 0; ks < 2; ++ks) {
            v16h af = load_frag_a(Aw, 64, m0, ks * 32, lane);
            v16h bf = load_frag_b(Bs, 64, n0, ks * 32, lane);
            acc = wmma16(af, bf, acc);
        }
        int ncol = lane & 15, hi = lane >> 4;
        int colg = col0 + n0 + ncol;
        int n = colg / T, t = colg % T;
        for (int r8 = 0; r8 < 8; ++r8) {
            int o = m0 + r8 + hi * 8;
            float vv = acc[r8] + cb[o];
            vv = vv > 0.f ? vv : 0.f;
            out[(((size_t)b * C + o) * NN + n) * T + t] = vv;
        }
    }
}

// ---------------- workspace layout (bytes) ----------------
constexpr size_t OFF_QUERY = 0;                                   // f32 B*C*NT
constexpr size_t OFF_VRT   = OFF_QUERY + (size_t)BB * C * NT * 4; // f16 B*CT*NN
constexpr size_t OFF_QST   = OFF_VRT   + (size_t)BB * CT * NN * 2;// f16 B*NN*C
constexpr size_t OFF_A1    = OFF_QST   + (size_t)BB * NN * C * 2; // f32 B*NN*NN
constexpr size_t OFF_A2    = OFF_A1    + (size_t)BB * NN * NN * 4;
constexpr size_t OFF_ADJH  = OFF_A2    + (size_t)BB * NN * NN * 4;// f16 B*NN*NN
constexpr size_t OFF_AGGH  = OFF_ADJH  + (size_t)BB * NN * NN * 2;// f16 B*NN*CT

extern "C" void kernel_launch(void* const* d_in, const int* in_sizes, int n_in,
                              void* d_out, int out_size, void* d_ws, size_t ws_size,
                              hipStream_t stream) {
    const float* x    = (const float*)d_in[0];
    const float* qw   = (const float*)d_in[1];
    const float* qb   = (const float*)d_in[2];
    const float* vw   = (const float*)d_in[3];
    const float* vb   = (const float*)d_in[4];
    const float* cw   = (const float*)d_in[5];
    const float* cb   = (const float*)d_in[6];
    const float* memf = (const float*)d_in[7];
    const float* fc1w = (const float*)d_in[8];
    const float* fc1b = (const float*)d_in[9];
    const float* fc2w = (const float*)d_in[10];
    const float* fc2b = (const float*)d_in[11];
    float* out = (float*)d_out;

    char* ws = (char*)d_ws;
    float*     query = (float*)(ws + OFF_QUERY);
    _Float16*  vrT   = (_Float16*)(ws + OFF_VRT);
    _Float16*  qsT   = (_Float16*)(ws + OFF_QST);
    float*     a1    = (float*)(ws + OFF_A1);
    float*     a2    = (float*)(ws + OFF_A2);
    _Float16*  adjh  = (_Float16*)(ws + OFF_ADJH);
    _Float16*  aggh  = (_Float16*)(ws + OFF_AGGH);

    k1_qv <<<BB * (NT / 64),      256, 0, stream>>>(x, qw, qb, vw, vb, query, vrT);
    k2_qs <<<BB * (NN / 4),       256, 0, stream>>>(query, qsT);
    k3_adj<<<2 * BB * 16 * 16,    256, 0, stream>>>(qsT, memf, a1, a2);
    k4_fuse<<<BB * NN,            256, 0, stream>>>(a1, a2, fc1w, fc1b, fc2w, fc2b, adjh);
    k6_agg<<<BB * 16 * 12,        256, 0, stream>>>(adjh, vrT, aggh);
    k7_out<<<BB * (NT / 64),      256, 0, stream>>>(aggh, cw, cb, out);
}
